// Transformer_3839700763128
// MI455X (gfx1250) — compile-verified, matrix operand order checked by reading, not run on hardware
//
#include <hip/hip_runtime.h>

// ============================================================================
// MI455X (gfx1250, wave32) implementation of the conv-frontend + 6-layer
// transformer encoder. All large GEMMs go through v_wmma_f32_16x16x32_bf16,
// with double-buffered LDS tiles filled by CDNA5 async global->LDS copies
// (GLOBAL_LOAD_ASYNC_TO_LDS_B128 / s_wait_asynccnt) so DMA overlaps the
// matrix pipes. Compute-bound workload (~340 GFLOP, AI >> 100 FLOP/B vs
// 23.3 TB/s HBM). Workspace requirement: ~138 MB.
// ============================================================================

typedef __attribute__((ext_vector_type(16))) __bf16 v16bf;
typedef __attribute__((ext_vector_type(8)))  float  v8f;

union BF16x16 { v16bf v; uint4 u[2]; };

__device__ __forceinline__ __bf16 f2bf(float f) {
  union { float f; unsigned u; } in; in.f = f;
  unsigned u = in.u;
  u += 0x7fffu + ((u >> 16) & 1u);        // round-to-nearest-even
  union { unsigned short s; __bf16 b; } o; o.s = (unsigned short)(u >> 16);
  return o.b;
}

template <typename T> __device__ __forceinline__ T cvt_out(float v);
template <> __device__ __forceinline__ float  cvt_out<float>(float v)  { return v; }
template <> __device__ __forceinline__ __bf16 cvt_out<__bf16>(float v) { return f2bf(v); }

// ---- CDNA5 async global->LDS copy (16B per lane), tracked by ASYNCcnt ------
__device__ __forceinline__ void async_copy16(const __bf16* lds_dst,
                                             const __bf16* gsrc) {
  unsigned int  lds = (unsigned int)(uintptr_t)lds_dst;   // low 32b = LDS offset
  unsigned long long ga = (unsigned long long)(uintptr_t)gsrc;
  asm volatile("global_load_async_to_lds_b128 %0, %1, off"
               :: "v"(lds), "v"(ga) : "memory");
}
__device__ __forceinline__ void wait_async0() {
#if __has_builtin(__builtin_amdgcn_s_wait_asynccnt)
  __builtin_amdgcn_s_wait_asynccnt(0);
#else
  asm volatile("s_wait_asynccnt 0" ::: "memory");
#endif
}

// ----------------------------------------------------------------------------
// f32 -> bf16 conversion (weights / activations)
// ----------------------------------------------------------------------------
__global__ void cvt_bf16_kernel(const float* __restrict__ src,
                                __bf16* __restrict__ dst, int n) {
  int i = blockIdx.x * 256 + threadIdx.x;
  if (i < n) dst[i] = f2bf(src[i]);
}

// ----------------------------------------------------------------------------
// WMMA GEMM: C(M,N) = A(M,K) @ W(N,K)^T + bias, A/W bf16 row-major.
// Block tile 128x64, BK=32, 256 threads = 8 waves (4x2 wave grid),
// each wave -> 32x32 output = 2x2 tiles of v_wmma_f32_16x16x32_bf16.
// Double-buffered LDS, tiles filled with async global->LDS DMA.
// M % 128 == 0, N % 64 == 0, K % 32 == 0 (true for every call site).
// ----------------------------------------------------------------------------
template <typename OutT, int ACT>
__global__ __launch_bounds__(256)
void gemm_bf16(const __bf16* __restrict__ A, const __bf16* __restrict__ W,
               const float* __restrict__ bias, OutT* __restrict__ C,
               int M, int N, int K) {
  constexpr int BM = 128, BN = 64, BK = 32;
  constexpr int LDA = 40, LDB = 40;           // bf16 elems; pad spreads LDS banks
  constexpr int BUFE = BM * LDA + BN * LDB;   // 7680 elems per buffer
  __shared__ __bf16 smem[2 * BUFE];           // 30 KB double buffer
  const int tid  = threadIdx.x;
  const int lane = tid & 31;
  const int wave = tid >> 5;
  const int wm   = wave & 3;                  // 0..3 -> 32-row group
  const int wn   = wave >> 2;                 // 0..1 -> 32-col group
  const int blockM = blockIdx.y * BM;
  const int blockN = blockIdx.x * BN;
  (void)M;

  // stage K-tile k0 into buffer p via async DMA (3 x 16B per thread)
  auto stage = [&](int p, int k0) {
    __bf16* sA = smem + p * BUFE;
    __bf16* sB = sA + BM * LDA;
#pragma unroll
    for (int i = 0; i < 2; ++i) {             // A: 128x32 = 512 16B chunks
      int chunk = i * 256 + tid;
      int r = chunk >> 2, c8 = (chunk & 3) * 8;
      async_copy16(&sA[r * LDA + c8], A + (size_t)(blockM + r) * K + k0 + c8);
    }
    {                                         // W: 64x32 = 256 16B chunks
      int r = tid >> 2, c8 = (tid & 3) * 8;
      async_copy16(&sB[r * LDB + c8], W + (size_t)(blockN + r) * K + k0 + c8);
    }
  };

  v8f acc[2][2] = {};
  const int rsel = lane & 15;
  const int koff = (lane >> 4) * 8;           // ISA 16-bit operand K striping

  stage(0, 0);
  wait_async0();
  __syncthreads();

  int p = 0;
  for (int k0 = 0; k0 < K; k0 += BK) {
    if (k0 + BK < K) stage(1 - p, k0 + BK);   // DMA next tile under the WMMAs

    const __bf16* cA = smem + p * BUFE;
    const __bf16* cB = cA + BM * LDA;
    BF16x16 a[2], b[2];
#pragma unroll
    for (int i = 0; i < 2; ++i) {
      int row = wm * 32 + i * 16 + rsel;
      a[i].u[0] = *(const uint4*)&cA[row * LDA + koff];
      a[i].u[1] = *(const uint4*)&cA[row * LDA + koff + 16];
    }
#pragma unroll
    for (int j = 0; j < 2; ++j) {
      int col = wn * 32 + j * 16 + rsel;
      b[j].u[0] = *(const uint4*)&cB[col * LDB + koff];
      b[j].u[1] = *(const uint4*)&cB[col * LDB + koff + 16];
    }
#pragma unroll
    for (int i = 0; i < 2; ++i)
#pragma unroll
      for (int j = 0; j < 2; ++j)
        acc[i][j] = __builtin_amdgcn_wmma_f32_16x16x32_bf16(
            false, a[i].v, false, b[j].v, (short)0, acc[i][j], false, false);

    wait_async0();                            // this wave's DMA into 1-p done
    __syncthreads();                          // everyone's DMA done, reads done
    p ^= 1;
  }

  // epilogue: C/D layout - lanes 0-15: N=lane, M=v; lanes 16-31: N=lane-16, M=v+8
  const int nsub  = lane & 15;
  const int mbase = (lane >> 4) * 8;
#pragma unroll
  for (int i = 0; i < 2; ++i)
#pragma unroll
    for (int j = 0; j < 2; ++j) {
      int n = blockN + wn * 32 + j * 16 + nsub;
      float bv = bias[n];
#pragma unroll
      for (int vv = 0; vv < 8; ++vv) {
        int m = blockM + wm * 32 + i * 16 + mbase + vv;
        float val = acc[i][j][vv] + bv;
        if (ACT == 1) val = fmaxf(val, 0.f);
        C[(size_t)m * N + n] = cvt_out<OutT>(val);
      }
    }
}

// ----------------------------------------------------------------------------
// Attention logits: per (head,batch): 64x64 = Q(64x64) @ K(64x64)^T, *1/sqrt(64)
// Q/K stored bf16 (B*64, 512) with feature = h*64+d. 128 threads = 4 waves,
// wave w -> query rows [16w,16w+16), K-dim = 2 WMMA steps.
// Writes f32 logits straight into d_out's attn slice.
// ----------------------------------------------------------------------------
__global__ __launch_bounds__(128)
void attn_scores(const __bf16* __restrict__ q, const __bf16* __restrict__ k,
                 float* __restrict__ out) {
  int hb = blockIdx.x;            // h*128 + b
  int h = hb >> 7, b = hb & 127;
  int lane = threadIdx.x & 31;
  int wave = threadIdx.x >> 5;
  int rsel = lane & 15;
  int koff = (lane >> 4) * 8;
  const size_t base = (size_t)b * 64 * 512 + h * 64;

  v8f acc[4] = {};
#pragma unroll
  for (int ks = 0; ks < 2; ++ks) {
    int kk = ks * 32;
    BF16x16 a;
    const __bf16* ap = q + base + (size_t)(wave * 16 + rsel) * 512 + kk + koff;
    a.u[0] = *(const uint4*)ap;
    a.u[1] = *(const uint4*)(ap + 16);
#pragma unroll
    for (int j = 0; j < 4; ++j) {
      BF16x16 bf;
      const __bf16* bp = k + base + (size_t)(j * 16 + rsel) * 512 + kk + koff;
      bf.u[0] = *(const uint4*)bp;
      bf.u[1] = *(const uint4*)(bp + 16);
      acc[j] = __builtin_amdgcn_wmma_f32_16x16x32_bf16(
          false, a.v, false, bf.v, (short)0, acc[j], false, false);
    }
  }
  float* o = out + (size_t)hb * 4096;
  int nsub = lane & 15, mbase = (lane >> 4) * 8;
#pragma unroll
  for (int j = 0; j < 4; ++j)
#pragma unroll
    for (int vv = 0; vv < 8; ++vv)
      o[(wave * 16 + mbase + vv) * 64 + j * 16 + nsub] = acc[j][vv] * 0.125f;
}

// ----------------------------------------------------------------------------
// BatchNorm stats over (h,b,k) per query channel q (64 channels, 65536 samples)
// stats[q] = mean, stats[64+q] = rsqrt(var+eps)
// ----------------------------------------------------------------------------
__global__ __launch_bounds__(256)
void bn_stats(const float* __restrict__ attn, float* __restrict__ stats) {
  int q = blockIdx.x;
  float s = 0.f, ss = 0.f;
  for (int hb = threadIdx.x; hb < 1024; hb += 256) {
    const float* p = attn + (size_t)hb * 4096 + q * 64;
    for (int kc = 0; kc < 64; ++kc) { float v = p[kc]; s += v; ss += v * v; }
  }
  __shared__ float rs[256], rq[256];
  rs[threadIdx.x] = s; rq[threadIdx.x] = ss;
  __syncthreads();
  for (int st = 128; st > 0; st >>= 1) {
    if (threadIdx.x < st) { rs[threadIdx.x] += rs[threadIdx.x + st];
                            rq[threadIdx.x] += rq[threadIdx.x + st]; }
    __syncthreads();
  }
  if (threadIdx.x == 0) {
    float m = rs[0] * (1.f / 65536.f);
    float var = rq[0] * (1.f / 65536.f) - m * m;
    stats[q] = m;
    stats[64 + q] = rsqrtf(var + 1e-5f);
  }
}

// ----------------------------------------------------------------------------
// BN (affine) + softmax over k, in place in d_out. 1 wave per row, 8 waves.
// ----------------------------------------------------------------------------
__global__ __launch_bounds__(256)
void bn_softmax(float* __restrict__ attn, const float* __restrict__ stats,
                const float* __restrict__ g, const float* __restrict__ bt) {
  int hb = blockIdx.x;
  int lane = threadIdx.x & 31;
  int wave = threadIdx.x >> 5;
  float* base = attn + (size_t)hb * 4096;
  for (int r = 0; r < 8; ++r) {
    int q = wave * 8 + r;
    float* p = base + q * 64;
    float sc = stats[64 + q] * g[q];
    float sh = bt[q] - stats[q] * sc;
    float t0 = p[lane] * sc + sh;
    float t1 = p[lane + 32] * sc + sh;
    float mx = fmaxf(t0, t1);
    for (int o = 16; o > 0; o >>= 1) mx = fmaxf(mx, __shfl_xor(mx, o, 32));
    float e0 = __expf(t0 - mx), e1 = __expf(t1 - mx);
    float s = e0 + e1;
    for (int o = 16; o > 0; o >>= 1) s += __shfl_xor(s, o, 32);
    float rinv = 1.f / s;
    p[lane] = e0 * rinv;
    p[lane + 32] = e1 * rinv;
  }
}

// ----------------------------------------------------------------------------
// out = P(64x64) @ V(64x64) per (h,b). V tile staged into LDS with contiguous
// b128 loads; the transposed per-lane gathers then hit LDS, not VMEM.
// Writes head-concatenated bf16 activations (B*64, 512).
// ----------------------------------------------------------------------------
__global__ __launch_bounds__(128)
void attn_v(const float* __restrict__ attn, const __bf16* __restrict__ v,
            __bf16* __restrict__ outcat) {
  constexpr int LDV = 72;                     // pad (144B rows, 16B aligned)
  __shared__ __bf16 sV[64 * LDV];
  int hb = blockIdx.x;
  int h = hb >> 7, b = hb & 127;
  int tid = threadIdx.x;
  int lane = tid & 31;
  int wave = tid >> 5;
  int rsel = lane & 15;
  int koff = (lane >> 4) * 8;
  const float* p = attn + (size_t)hb * 4096;

  // stage V tile: rows = key index, cols = head feature. 512 x 16B chunks.
#pragma unroll
  for (int i = 0; i < 4; ++i) {
    int chunk = i * 128 + tid;
    int r = chunk >> 3, c8 = (chunk & 7) * 8;
    *(uint4*)&sV[r * LDV + c8] =
        *(const uint4*)(v + ((size_t)b * 64 + r) * 512 + h * 64 + c8);
  }
  __syncthreads();

  v8f acc[4] = {};
#pragma unroll
  for (int ks = 0; ks < 2; ++ks) {
    int kk = ks * 32;
    v16bf av;
    {
      const float* row = p + (wave * 16 + rsel) * 64 + kk + koff;
#pragma unroll
      for (int e = 0; e < 8; ++e) { av[e] = f2bf(row[e]); av[8 + e] = f2bf(row[16 + e]); }
    }
#pragma unroll
    for (int j = 0; j < 4; ++j) {
      v16bf bv;
      int n = j * 16 + rsel;
#pragma unroll
      for (int e = 0; e < 8; ++e) {
        bv[e]     = sV[(kk + koff + e) * LDV + n];
        bv[8 + e] = sV[(kk + koff + 16 + e) * LDV + n];
      }
      acc[j] = __builtin_amdgcn_wmma_f32_16x16x32_bf16(
          false, av, false, bv, (short)0, acc[j], false, false);
    }
  }
  int nsub = lane & 15, mbase = (lane >> 4) * 8;
#pragma unroll
  for (int j = 0; j < 4; ++j)
#pragma unroll
    for (int vv = 0; vv < 8; ++vv) {
      int qrow = wave * 16 + mbase + vv;
      outcat[((size_t)b * 64 + qrow) * 512 + h * 64 + j * 16 + nsub] =
          f2bf(acc[j][vv]);
    }
}

// ----------------------------------------------------------------------------
// LayerNorm(a + resid) over 512 feats; 1 wave per row (16 values / lane).
// Emits both f32 (for residual chain) and bf16 (for next GEMM).
// ----------------------------------------------------------------------------
__global__ __launch_bounds__(256)
void layernorm_add(const float* __restrict__ a, const float* __restrict__ resid,
                   const float* __restrict__ g, const float* __restrict__ bt,
                   float* __restrict__ of, __bf16* __restrict__ ob) {
  int row = blockIdx.x * 8 + (threadIdx.x >> 5);
  int lane = threadIdx.x & 31;
  const float* pa = a + (size_t)row * 512;
  const float* pr = resid + (size_t)row * 512;
  float v[16]; float s = 0.f;
#pragma unroll
  for (int i = 0; i < 16; ++i) { v[i] = pa[lane + 32 * i] + pr[lane + 32 * i]; s += v[i]; }
  for (int o = 16; o > 0; o >>= 1) s += __shfl_xor(s, o, 32);
  float m = s * (1.f / 512.f);
  float qv = 0.f;
#pragma unroll
  for (int i = 0; i < 16; ++i) { float d = v[i] - m; qv += d * d; }
  for (int o = 16; o > 0; o >>= 1) qv += __shfl_xor(qv, o, 32);
  float r = rsqrtf(qv * (1.f / 512.f) + 1e-5f);
#pragma unroll
  for (int i = 0; i < 16; ++i) {
    int c = lane + 32 * i;
    float out = (v[i] - m) * r * g[c] + bt[c];
    of[(size_t)row * 512 + c] = out;
    ob[(size_t)row * 512 + c] = f2bf(out);
  }
}

// ----------------------------------------------------------------------------
// Conv1d (+bias) fused with maxpool(2). One thread per output element.
// ----------------------------------------------------------------------------
__global__ void conv_pool(const float* __restrict__ x, const float* __restrict__ w,
                          const float* __restrict__ bias, float* __restrict__ y,
                          int Bn, int Cin, int Lin, int Cout, int Kk, int pad) {
  int Lout = (Lin + 2 * pad - Kk + 1) >> 1;
  int idx = blockIdx.x * 256 + threadIdx.x;
  if (idx >= Bn * Cout * Lout) return;
  int lo = idx % Lout;
  int co = (idx / Lout) % Cout;
  int b  = idx / (Lout * Cout);
  float best = -3.4e38f;
  for (int t = 0; t < 2; ++t) {
    int pos = 2 * lo + t;
    float acc = bias[co];
    for (int ci = 0; ci < Cin; ++ci) {
      const float* xp = x + ((size_t)b * Cin + ci) * Lin;
      const float* wp = w + ((size_t)co * Cin + ci) * Kk;
      for (int kk = 0; kk < Kk; ++kk) {
        int ip = pos + kk - pad;
        if (ip >= 0 && ip < Lin) acc += wp[kk] * xp[ip];
      }
    }
    best = fmaxf(best, acc);
  }
  y[idx] = best;
}

// ---------------------------- SE block ---------------------------------------
__global__ void se_mean(const float* __restrict__ s1, const float* __restrict__ s2,
                        const float* __restrict__ s3, float* __restrict__ out) {
  int idx = blockIdx.x * 256 + threadIdx.x;       // b*1536 + c
  if (idx >= 128 * 1536) return;
  int c = idx % 1536, b = idx / 1536;
  const float* s = (c < 512) ? s1 : (c < 1024 ? s2 : s3);
  const float* p = s + ((size_t)b * 512 + (c & 511)) * 64;
  float acc = 0.f;
  for (int l = 0; l < 64; ++l) acc += p[l];
  out[idx] = acc * (1.f / 64.f);
}

__global__ void se_fc1(const float* __restrict__ mean, const float* __restrict__ w1,
                       float* __restrict__ h) {
  int idx = blockIdx.x * 256 + threadIdx.x;       // b*32 + j
  if (idx >= 128 * 32) return;
  int j = idx & 31, b = idx >> 5;
  const float* wp = w1 + (size_t)j * 1536;
  const float* mp = mean + (size_t)b * 1536;
  float acc = 0.f;
  for (int c = 0; c < 1536; ++c) acc += wp[c] * mp[c];
  h[idx] = fmaxf(acc, 0.f);
}

__global__ void se_fc2(const float* __restrict__ h, const float* __restrict__ w2,
                       float* __restrict__ se) {
  int idx = blockIdx.x * 256 + threadIdx.x;       // b*1536 + c
  if (idx >= 128 * 1536) return;
  int c = idx % 1536, b = idx / 1536;
  const float* wp = w2 + (size_t)c * 32;
  const float* hp = h + (size_t)b * 32;
  float acc = 0.f;
  for (int j = 0; j < 32; ++j) acc += wp[j] * hp[j];
  se[idx] = 1.f / (1.f + __expf(-acc));
}

// SE-scale + (B,C,L) -> token-major bf16 pack for the c7 GEMM
__global__ void pack_scaled(const float* __restrict__ s1, const float* __restrict__ s2,
                            const float* __restrict__ s3, const float* __restrict__ se,
                            __bf16* __restrict__ S) {
  int idx = blockIdx.x * 256 + threadIdx.x;       // (b*64+l)*1536 + c
  if (idx >= 8192 * 1536) return;
  int c = idx % 1536;
  int bl = idx / 1536;
  int l = bl & 63, b = bl >> 6;
  const float* s = (c < 512) ? s1 : (c < 1024 ? s2 : s3);
  float v = s[((size_t)b * 512 + (c & 511)) * 64 + l] * se[b * 1536 + c];
  S[idx] = f2bf(v);
}

// x_t0 = c7_gemm_out + res(transposed); enc = x_t0 + sinusoid PE (pos = l+1)
__global__ void add_res_pos(const float* __restrict__ gout, const float* __restrict__ res,
                            float* __restrict__ xt0, float* __restrict__ encf,
                            __bf16* __restrict__ encb) {
  int idx = blockIdx.x * 256 + threadIdx.x;       // (b*64+l)*512 + d
  if (idx >= 8192 * 512) return;
  int d = idx & 511;
  int bl = idx >> 9;
  int l = bl & 63, b = bl >> 6;
  float x = gout[idx] + res[((size_t)b * 512 + d) * 64 + l];
  xt0[idx] = x;
  float pos = (float)(l + 1);
  float inv = __expf(-(float)(2 * (d >> 1)) * 0.001953125f * 9.210340372f);
  float ang = pos * inv;
  float pe = (d & 1) ? __cosf(ang) : __sinf(ang);
  float e = x + pe;
  encf[idx] = e;
  encb[idx] = f2bf(e);
}

// out[b] = sum_d lin_w[d] * (sum_l x_t0[b,l,d]*cov_w[l] + cov_b) + lin_b
__global__ __launch_bounds__(256)
void final_out(const float* __restrict__ xt0, const float* __restrict__ cov_w,
               const float* __restrict__ cov_b, const float* __restrict__ lin_w,
               const float* __restrict__ lin_b, float* __restrict__ out) {
  int b = blockIdx.x;
  float part = 0.f;
  for (int d = threadIdx.x; d < 512; d += 256) {
    float acc = cov_b[0];
    const float* p = xt0 + (size_t)b * 64 * 512 + d;
    for (int l = 0; l < 64; ++l) acc += p[(size_t)l * 512] * cov_w[l];
    part += acc * lin_w[d];
  }
  __shared__ float red[256];
  red[threadIdx.x] = part;
  __syncthreads();
  for (int st = 128; st > 0; st >>= 1) {
    if (threadIdx.x < st) red[threadIdx.x] += red[threadIdx.x + st];
    __syncthreads();
  }
  if (threadIdx.x == 0) out[b] = red[0] + lin_b[0];
}

// ============================================================================
// Host orchestration
// ============================================================================
extern "C" void kernel_launch(void* const* d_in, const int* in_sizes, int n_in,
                              void* d_out, int out_size, void* d_ws, size_t ws_size,
                              hipStream_t stream) {
  (void)in_sizes; (void)out_size; (void)ws_size;
  if (n_in < 25 + 6 * 18) return;
  auto F = [&](int i) { return (const float*)d_in[i]; };
  const float *src   = F(0);
  const float *w_emb0 = F(1),  *b_emb0 = F(2);
  const float *w_c2   = F(3),  *b_c2   = F(4);
  const float *w_emb1 = F(5),  *b_emb1 = F(6);
  const float *w_c4   = F(7),  *b_c4   = F(8);
  const float *w_emb2 = F(9),  *b_emb2 = F(10);
  const float *w_c6   = F(11), *b_c6   = F(12);
  const float *w_c7   = F(13), *b_c7   = F(14);
  const float *w_res1 = F(15), *b_res1 = F(16);
  const float *w_res2 = F(17), *b_res2 = F(18);
  const float *se_w1  = F(19), *se_w2  = F(20);
  const float *cov_w  = F(21), *cov_b  = F(22);
  const float *lin_w  = F(23), *lin_b  = F(24);

  char* ws = (char*)d_ws;
  float*  XT0  = (float*)ws;                          // (8192,512) f32, permanent
  char*   pool = ws + (size_t)16777216;
  // --- phase A (conv frontend) layout ---
  float*  T1A = (float*)(pool + 0);
  float*  T1B = (float*)(pool + 2097152);
  float*  T1C = (float*)(pool + 4194304);
  float*  T1R = (float*)(pool + 6291456);
  float*  S2A = (float*)(pool + 8388608);
  float*  S2B = (float*)(pool + 25165824);
  float*  S2C = (float*)(pool + 41943040);
  float*  S2R = (float*)(pool + 58720256);
  float*  SEM = (float*)(pool + 75497472);
  float*  SEH = (float*)(pool + 76283904);
  float*  SES = (float*)(pool + 76300288);
  __bf16* SBF = (__bf16*)(pool + 77086720);
  __bf16* WC7 = (__bf16*)(pool + 102252544);
  float*  GOUT= (float*)(pool + 103825408);
  // --- phase B (encoder layers) layout, overlaps dead phase-A scratch ---
  float*  ENCF= (float*)(pool + 0);
  __bf16* ENCB= (__bf16*)(pool + 16777216);
  __bf16* QB  = (__bf16*)(pool + 25165824);
  __bf16* KB  = (__bf16*)(pool + 33554432);
  __bf16* VB  = (__bf16*)(pool + 41943040);
  __bf16* ATB = (__bf16*)(pool + 50331648);
  float*  PROJ= (float*)(pool + 58720256);
  __bf16* HB  = (__bf16*)(pool + 75497472);
  __bf16* WB  = (__bf16*)(pool + 109051904);
  float*  BNST= (float*)(pool + 111149056);

  // ---- conv frontend ----
  int n1 = 128 * 32 * 128;
  conv_pool<<<n1 / 256, 256, 0, stream>>>(src, w_emb0, b_emb0, T1A, 128, 1, 256, 32, 15, 7);
  conv_pool<<<n1 / 256, 256, 0, stream>>>(src, w_emb1, b_emb1, T1B, 128, 1, 256, 32, 13, 6);
  conv_pool<<<n1 / 256, 256, 0, stream>>>(src, w_emb2, b_emb2, T1C, 128, 1, 256, 32, 11, 5);
  conv_pool<<<n1 / 256, 256, 0, stream>>>(src, w_res1, b_res1, T1R, 128, 1, 256, 32, 1, 0);
  int n2 = 128 * 512 * 64;
  conv_pool<<<n2 / 256, 256, 0, stream>>>(T1A, w_c2,   b_c2,   S2A, 128, 32, 128, 512, 9, 4);
  conv_pool<<<n2 / 256, 256, 0, stream>>>(T1B, w_c4,   b_c4,   S2B, 128, 32, 128, 512, 7, 3);
  conv_pool<<<n2 / 256, 256, 0, stream>>>(T1C, w_c6,   b_c6,   S2C, 128, 32, 128, 512, 5, 2);
  conv_pool<<<n2 / 256, 256, 0, stream>>>(T1R, w_res2, b_res2, S2R, 128, 32, 128, 512, 1, 0);

  se_mean<<<(128 * 1536) / 256, 256, 0, stream>>>(S2A, S2B, S2C, SEM);
  se_fc1 <<<(128 * 32) / 256,   256, 0, stream>>>(SEM, se_w1, SEH);
  se_fc2 <<<(128 * 1536) / 256, 256, 0, stream>>>(SEH, se_w2, SES);
  pack_scaled<<<(8192 * 1536) / 256, 256, 0, stream>>>(S2A, S2B, S2C, SES, SBF);

  cvt_bf16_kernel<<<(512 * 1536) / 256, 256, 0, stream>>>(w_c7, WC7, 512 * 1536);
  gemm_bf16<float, 0><<<dim3(8, 64), 256, 0, stream>>>(SBF, WC7, b_c7, GOUT, 8192, 512, 1536);
  add_res_pos<<<(8192 * 512) / 256, 256, 0, stream>>>(GOUT, S2R, XT0, ENCF, ENCB);

  // ---- encoder layers ----
  const dim3 g512(8, 64);      // N=512 GEMMs
  const dim3 g2048(32, 64);    // N=2048 GEMM
  for (int i = 0; i < 6; ++i) {
    int base = 25 + i * 18;
    const float *wq = F(base + 0),  *bq = F(base + 1);
    const float *wk = F(base + 2),  *bk = F(base + 3);
    const float *wv = F(base + 4),  *bv = F(base + 5);
    const float *fcw = F(base + 6), *fcb = F(base + 7);
    const float *bng = F(base + 8), *bnb = F(base + 9);
    const float *l1g = F(base + 10), *l1b = F(base + 11);
    const float *w1 = F(base + 12), *b1 = F(base + 13);
    const float *w2 = F(base + 14), *b2 = F(base + 15);
    const float *l2g = F(base + 16), *l2b = F(base + 17);
    float* ATT = (float*)d_out + 128 + (size_t)i * 4194304;

    cvt_bf16_kernel<<<1024, 256, 0, stream>>>(wq, WB, 262144);
    gemm_bf16<__bf16, 0><<<g512, 256, 0, stream>>>(ENCB, WB, bq, QB, 8192, 512, 512);
    cvt_bf16_kernel<<<1024, 256, 0, stream>>>(wk, WB, 262144);
    gemm_bf16<__bf16, 0><<<g512, 256, 0, stream>>>(ENCB, WB, bk, KB, 8192, 512, 512);
    cvt_bf16_kernel<<<1024, 256, 0, stream>>>(wv, WB, 262144);
    gemm_bf16<__bf16, 0><<<g512, 256, 0, stream>>>(ENCB, WB, bv, VB, 8192, 512, 512);

    attn_scores<<<1024, 128, 0, stream>>>(QB, KB, ATT);
    bn_stats<<<64, 256, 0, stream>>>(ATT, BNST);
    bn_softmax<<<1024, 256, 0, stream>>>(ATT, BNST, bng, bnb);
    attn_v<<<1024, 128, 0, stream>>>(ATT, VB, ATB);

    cvt_bf16_kernel<<<1024, 256, 0, stream>>>(fcw, WB, 262144);
    gemm_bf16<float, 0><<<g512, 256, 0, stream>>>(ATB, WB, fcb, PROJ, 8192, 512, 512);
    layernorm_add<<<1024, 256, 0, stream>>>(PROJ, ENCF, l1g, l1b, ENCF, ENCB);

    cvt_bf16_kernel<<<4096, 256, 0, stream>>>(w1, WB, 1048576);
    gemm_bf16<__bf16, 1><<<g2048, 256, 0, stream>>>(ENCB, WB, b1, HB, 8192, 2048, 512);
    cvt_bf16_kernel<<<4096, 256, 0, stream>>>(w2, WB, 1048576);
    gemm_bf16<float, 0><<<g512, 256, 0, stream>>>(HB, WB, b2, PROJ, 8192, 512, 2048);
    layernorm_add<<<1024, 256, 0, stream>>>(PROJ, ENCF, l2g, l2b, ENCF, ENCB);
  }

  // ---- output head (uses pre-encoder x_t) ----
  final_out<<<128, 256, 0, stream>>>(XT0, cov_w, cov_b, lin_w, lin_b, (float*)d_out);
}
